// MQA_60327110639653
// MI455X (gfx1250) — compile-verified, matrix operand order checked by reading, not run on hardware
//
#include <hip/hip_runtime.h>

// ---------------------------------------------------------------------------
// MQA forward for MI455X (gfx1250), wave32 WMMA bf16 everywhere.
//   1) x -> bf16; W{q,k,v,o} -> transposed bf16 (Wt[n][k])  [one-time prep]
//   2) Q = (x@Wq+bq)*0.125 -> bf16 [M,1024]
//      K =  x@Wk+bk        -> bf16 [M,64]
//      Vt = (x@Wv+bv)^T    -> bf16 [64,M]
//   3) flash attention: S^T = K.Q^T (lane owns one query's scores ->
//      in-lane softmax reductions, probs already in PV A-frag layout)
//   4) out = ctx@Wo+bo -> f32
// ---------------------------------------------------------------------------

#define D_MODEL 1024
#define NUM_HEADS 16
#define D_K 64
#define BATCH 4
#define SEQ 2048
#define MTOT (BATCH * SEQ)  // 8192

typedef __attribute__((ext_vector_type(16))) __bf16 v16bf;
typedef __attribute__((ext_vector_type(8)))  float  v8f;

__device__ __forceinline__ v8f wmma_bf16(v16bf a, v16bf b, v8f c) {
  // (neg_a, A, neg_b, B, c_mod, C, reuse_a, reuse_b)
  return __builtin_amdgcn_wmma_f32_16x16x32_bf16(false, a, false, b,
                                                 (short)0, c, false, false);
}

// Fragment (A or B) from bf16 memory: lane holds k = {kbase..kbase+7,
// kbase+16..kbase+23} of its row/column, kbase=0|8 by lane half. Caller
// passes base = &row[kbase]; both halves are 16B -> two b128 loads.
__device__ __forceinline__ v16bf load_frag_bf16(const __bf16* base) {
  v16bf f;
#pragma unroll
  for (int i = 0; i < 8; ++i) {
    f[i]     = base[i];
    f[i + 8] = base[16 + i];
  }
  return f;
}

// ---------------------------------------------------------------------------
// Prep kernels (run once, tiny cost)
// ---------------------------------------------------------------------------
__global__ __launch_bounds__(256) void convert_bf16_kernel(
    const float* __restrict__ in, __bf16* __restrict__ out, int n) {
  int i = (blockIdx.x * 256 + threadIdx.x) * 4;
  if (i < n) {
    float4 v = *(const float4*)(in + i);
    out[i + 0] = (__bf16)v.x;
    out[i + 1] = (__bf16)v.y;
    out[i + 2] = (__bf16)v.z;
    out[i + 3] = (__bf16)v.w;
  }
}

// Wt[n][k] = bf16(W[k][n]); 32x32 LDS tile transpose.
__global__ __launch_bounds__(256) void transpose_bf16_kernel(
    const float* __restrict__ in, __bf16* __restrict__ out, int K, int N) {
  __shared__ float tile[32][33];
  const int k0 = blockIdx.x * 32, n0 = blockIdx.y * 32;
  const int c = threadIdx.x & 31, r8 = threadIdx.x >> 5;  // r8: 0..7
#pragma unroll
  for (int i = 0; i < 4; ++i) {
    int r = r8 + i * 8;
    tile[r][c] = in[(size_t)(k0 + r) * N + n0 + c];
  }
  __syncthreads();
#pragma unroll
  for (int i = 0; i < 4; ++i) {
    int r = r8 + i * 8;
    out[(size_t)(n0 + r) * K + k0 + c] = (__bf16)tile[c][r];
  }
}

// ---------------------------------------------------------------------------
// GEMM: C[M,N] = A[M,Kd](bf16) * Wt[N,Kd](bf16)^T + bias, epilogue scale.
// Block = 128 (4 waves); each wave: 32(M) x 64(N) tile, 8 f32 accumulators.
// Two A fragments share each B fragment -> 8 WMMAs per 12 b128 loads.
// All fragments are contiguous bf16 global_load_b128 (no LDS, no barriers).
// ---------------------------------------------------------------------------
template <int Kd, int N, bool TRANS_OUT, bool BF16_OUT>
__global__ __launch_bounds__(128) void gemm_bf16_kernel(
    const __bf16* __restrict__ A, const __bf16* __restrict__ Wt,
    const float* __restrict__ bias, void* __restrict__ Cout, float out_scale) {
  const int tid   = threadIdx.x;
  const int wave  = tid >> 5;
  const int lane  = tid & 31;
  const int l15   = lane & 15;
  const int hi    = lane >> 4;
  const int kbase = hi ? 8 : 0;

  const int m0 = blockIdx.x * 128 + wave * 32;
  const int n0 = blockIdx.y * 64;

  const __bf16* Arow0 = A + (size_t)(m0 + l15) * Kd + kbase;
  const __bf16* Arow1 = A + (size_t)(m0 + 16 + l15) * Kd + kbase;
  const __bf16* Wrow[4];
#pragma unroll
  for (int j = 0; j < 4; ++j)
    Wrow[j] = Wt + (size_t)(n0 + j * 16 + l15) * Kd + kbase;

  v8f acc0[4] = {}, acc1[4] = {};
  for (int k0 = 0; k0 < Kd; k0 += 32) {
    v16bf af0 = load_frag_bf16(Arow0 + k0);
    v16bf af1 = load_frag_bf16(Arow1 + k0);
#pragma unroll
    for (int j = 0; j < 4; ++j) {
      v16bf bf = load_frag_bf16(Wrow[j] + k0);
      acc0[j] = wmma_bf16(af0, bf, acc0[j]);
      acc1[j] = wmma_bf16(af1, bf, acc1[j]);
    }
  }

  // C/D layout: VGPR r -> (mtile + r + 8*hi, n0 + j*16 + l15)
#pragma unroll
  for (int j = 0; j < 4; ++j) {
    int   n  = n0 + j * 16 + l15;
    float bv = bias[n];
#pragma unroll
    for (int r = 0; r < 8; ++r) {
      int    m0r  = m0 + r + hi * 8;
      int    m1r  = m0 + 16 + r + hi * 8;
      float  val0 = (acc0[j][r] + bv) * out_scale;
      float  val1 = (acc1[j][r] + bv) * out_scale;
      size_t i0 = TRANS_OUT ? ((size_t)n * MTOT + m0r) : ((size_t)m0r * N + n);
      size_t i1 = TRANS_OUT ? ((size_t)n * MTOT + m1r) : ((size_t)m1r * N + n);
      if constexpr (BF16_OUT) {
        ((__bf16*)Cout)[i0] = (__bf16)val0;
        ((__bf16*)Cout)[i1] = (__bf16)val1;
      } else {
        ((float*)Cout)[i0] = val0;
        ((float*)Cout)[i1] = val1;
      }
    }
  }
}

// ---------------------------------------------------------------------------
// Flash attention (MQA): each wave owns a 16-query tile of one (b,h).
// Scores computed TRANSPOSED: S^T[key][q] = K_tile . Q_tile^T (A = K rows,
// B = Q rows; identical fragment bytes, swapped operands). Consequences:
//  * lane (n = lane&15) owns query q: its 8+8 accumulator regs hold keys
//    {0..7,16..23} (low half) / {8..15,24..31} (high half) of the 32-key
//    tile; the xor-16 partner holds the other 16 -> softmax max/sum are
//    in-lane reductions + ONE shfl_xor(16) each.
//  * exp'd probs are ALREADY in the PV A-fragment layout -> no LDS bounce.
//  * running m/l are one scalar per lane; only the o-rescale factor needs a
//    16-float LDS broadcast per iteration (C-layout rows are m-indexed).
// K frags loaded up-front (clause-batched), V frags loaded before the
// softmax so exp VALU hides their latency. Q pre-scaled by 1/sqrt(64).
// ---------------------------------------------------------------------------
#define ATT_WAVES 4

__global__ __launch_bounds__(128) void mqa_attn_kernel(
    const __bf16* __restrict__ Q, const __bf16* __restrict__ Kp,
    const __bf16* __restrict__ Vt, __bf16* __restrict__ ctx) {
  __shared__ __align__(16) float cl_all[ATT_WAVES * 16];

  const int tid   = threadIdx.x;
  const int wave  = tid >> 5;
  const int lane  = tid & 31;
  const int l15   = lane & 15;
  const int hi    = lane >> 4;
  const int kbase = hi ? 8 : 0;
  float* cl = cl_all + wave * 16;

  const int bh = blockIdx.y;  // 0..63
  const int b  = bh >> 4;
  const int h  = bh & 15;
  const int q0 = (blockIdx.x * ATT_WAVES + wave) * 16;

  // Q fragments (used as B operands: Bt[n=q][k=d] = Q row q)
  const __bf16* qrow =
      Q + (size_t)(b * SEQ + q0 + l15) * D_MODEL + h * D_K + kbase;
  v16bf qf0 = load_frag_bf16(qrow);
  v16bf qf1 = load_frag_bf16(qrow + 32);

  const __bf16* Kb = Kp + (size_t)b * SEQ * D_K;
  const __bf16* Vrow[4];
#pragma unroll
  for (int j = 0; j < 4; ++j)
    Vrow[j] = Vt + (size_t)(j * 16 + l15) * MTOT + (size_t)b * SEQ + kbase;

  float mrun = -1e30f, lrun = 0.0f;  // per-lane scalars (this lane's query)
  v8f o[4] = {};

  for (int kt = 0; kt < SEQ; kt += 32) {
    // ---- issue ALL K-fragment loads first (one clause), then 4 WMMAs with
    //      interleaved accumulation chains for ILP.
    const __bf16* kr0 = Kb + (size_t)(kt + l15) * D_K + kbase;
    const __bf16* kr1 = Kb + (size_t)(kt + 16 + l15) * D_K + kbase;
    v16bf kf00 = load_frag_bf16(kr0);
    v16bf kf01 = load_frag_bf16(kr0 + 32);
    v16bf kf10 = load_frag_bf16(kr1);
    v16bf kf11 = load_frag_bf16(kr1 + 32);

    v8f st0 = {}, st1 = {};
    st0 = wmma_bf16(kf00, qf0, st0);
    st1 = wmma_bf16(kf10, qf0, st1);
    st0 = wmma_bf16(kf01, qf1, st0);
    st1 = wmma_bf16(kf11, qf1, st1);

    // ---- V fragments early: softmax VALU below hides their latency
    v16bf vf[4];
#pragma unroll
    for (int j = 0; j < 4; ++j) vf[j] = load_frag_bf16(Vrow[j] + kt);

    // ---- in-lane max over this lane's 16 keys, + xor-16 partner
    float vmax = fmaxf(st0[0], st1[0]);
#pragma unroll
    for (int r = 1; r < 8; ++r) vmax = fmaxf(vmax, fmaxf(st0[r], st1[r]));
    vmax = fmaxf(vmax, __shfl_xor(vmax, 16, 32));

    float mnew = fmaxf(mrun, vmax);
    float corr = __expf(mrun - mnew);
    mrun = mnew;

    // ---- exp + in-lane sum (+ partner), probs straight into A-frag layout
    v16bf pf;
    float rs = 0.0f;
#pragma unroll
    for (int r = 0; r < 8; ++r) {
      float e0 = __expf(st0[r] - mnew);
      float e1 = __expf(st1[r] - mnew);
      rs += e0 + e1;
      pf[r]     = (__bf16)e0;  // keys kbase+r      of this query
      pf[r + 8] = (__bf16)e1;  // keys kbase+16+r   of this query
    }
    rs += __shfl_xor(rs, 16, 32);
    lrun = lrun * corr + rs;

    // ---- broadcast per-query corr to o's m-indexed rows (16 floats/wave)
    if (lane < 16) cl[lane] = corr;
    asm volatile("s_wait_dscnt 0" ::: "memory");
    float4 c0 = *(const float4*)(cl + hi * 8);
    float4 c1 = *(const float4*)(cl + hi * 8 + 4);
    float cc[8] = {c0.x, c0.y, c0.z, c0.w, c1.x, c1.y, c1.z, c1.w};
#pragma unroll
    for (int j = 0; j < 4; ++j)
#pragma unroll
      for (int r = 0; r < 8; ++r) o[j][r] *= cc[r];

    // ---- PV: o += P.V ; A = probs (in regs), B = Vt rows. 4 WMMAs.
#pragma unroll
    for (int j = 0; j < 4; ++j) o[j] = wmma_bf16(pf, vf[j], o[j]);
  }

  // ---- broadcast 1/l per query to m-indexed rows, normalize, store bf16
  {
    float inv = 1.0f / lrun;
    if (lane < 16) cl[lane] = inv;
  }
  asm volatile("s_wait_dscnt 0" ::: "memory");
  float4 c0 = *(const float4*)(cl + hi * 8);
  float4 c1 = *(const float4*)(cl + hi * 8 + 4);
  float cc[8] = {c0.x, c0.y, c0.z, c0.w, c1.x, c1.y, c1.z, c1.w};
#pragma unroll
  for (int j = 0; j < 4; ++j) {
    int d = h * D_K + j * 16 + l15;
#pragma unroll
    for (int r = 0; r < 8; ++r) {
      int m = q0 + r + hi * 8;
      ctx[(size_t)(b * SEQ + m) * D_MODEL + d] = (__bf16)(o[j][r] * cc[r]);
    }
  }
}

// ---------------------------------------------------------------------------
extern "C" void kernel_launch(void* const* d_in, const int* in_sizes, int n_in,
                              void* d_out, int out_size, void* d_ws,
                              size_t ws_size, hipStream_t stream) {
  const float* x  = (const float*)d_in[0];
  const float* Wq = (const float*)d_in[1];
  const float* bq = (const float*)d_in[2];
  const float* Wk = (const float*)d_in[3];
  const float* bk = (const float*)d_in[4];
  const float* Wv = (const float*)d_in[5];
  const float* bv = (const float*)d_in[6];
  const float* Wo = (const float*)d_in[7];
  const float* bo = (const float*)d_in[8];
  float* out = (float*)d_out;

  // Workspace layout (bf16 buffers)
  char*   ws    = (char*)d_ws;
  __bf16* xbf   = (__bf16*)ws;                      // M x 1024
  __bf16* Wqt   = xbf + (size_t)MTOT * D_MODEL;     // 1024 x 1024
  __bf16* Wkt   = Wqt + (size_t)D_MODEL * D_MODEL;  // 64 x 1024
  __bf16* Wvt   = Wkt + (size_t)D_K * D_MODEL;      // 64 x 1024
  __bf16* Wot   = Wvt + (size_t)D_K * D_MODEL;      // 1024 x 1024
  __bf16* Qbf   = Wot + (size_t)D_MODEL * D_MODEL;  // M x 1024
  __bf16* Kbf   = Qbf + (size_t)MTOT * D_MODEL;     // M x 64
  __bf16* Vtbf  = Kbf + (size_t)MTOT * D_K;         // 64 x M
  __bf16* ctxbf = Vtbf + (size_t)D_K * MTOT;        // M x 1024

  // ---- prep: converts + weight transposes
  {
    int n = MTOT * D_MODEL;
    convert_bf16_kernel<<<dim3(n / 1024), dim3(256), 0, stream>>>(x, xbf, n);
  }
  transpose_bf16_kernel<<<dim3(D_MODEL / 32, D_MODEL / 32), dim3(256), 0,
                          stream>>>(Wq, Wqt, D_MODEL, D_MODEL);
  transpose_bf16_kernel<<<dim3(D_MODEL / 32, D_K / 32), dim3(256), 0,
                          stream>>>(Wk, Wkt, D_MODEL, D_K);
  transpose_bf16_kernel<<<dim3(D_MODEL / 32, D_K / 32), dim3(256), 0,
                          stream>>>(Wv, Wvt, D_MODEL, D_K);
  transpose_bf16_kernel<<<dim3(D_MODEL / 32, D_MODEL / 32), dim3(256), 0,
                          stream>>>(Wo, Wot, D_MODEL, D_MODEL);

  dim3 blk(128);
  // Q = (x@Wq + bq) * 1/sqrt(64)  -> bf16
  gemm_bf16_kernel<D_MODEL, D_MODEL, false, true>
      <<<dim3(MTOT / 128, D_MODEL / 64), blk, 0, stream>>>(xbf, Wqt, bq, Qbf,
                                                           0.125f);
  // K = x@Wk + bk -> bf16
  gemm_bf16_kernel<D_MODEL, D_K, false, true>
      <<<dim3(MTOT / 128, 1), blk, 0, stream>>>(xbf, Wkt, bk, Kbf, 1.0f);
  // V^T = (x@Wv + bv)^T -> bf16 [64, M]
  gemm_bf16_kernel<D_MODEL, D_K, true, true>
      <<<dim3(MTOT / 128, 1), blk, 0, stream>>>(xbf, Wvt, bv, Vtbf, 1.0f);
  // attention -> ctx bf16
  mqa_attn_kernel<<<dim3(SEQ / (16 * ATT_WAVES), BATCH * NUM_HEADS), blk, 0,
                    stream>>>(Qbf, Kbf, Vtbf, ctxbf);
  // out = ctx@Wo + bo -> f32
  gemm_bf16_kernel<D_MODEL, D_MODEL, false, false>
      <<<dim3(MTOT / 128, D_MODEL / 64), blk, 0, stream>>>(ctxbf, Wot, bo, out,
                                                           1.0f);
}
